// Model_MultiHeadAttention_LSTM_3796751089882
// MI455X (gfx1250) — matrix operand order repaired and checked by reading, not compiled, not run on hardware
//
#include <hip/hip_runtime.h>
#include <hip/hip_bf16.h>
#include <math.h>

// ---------- constants ----------
#define Bsz   1024
#define Tlen  64
#define ENC_H 64
#define DEC_H 128
#define PRED  50
#define NHEAD 4
#define HD    32
#define LCELL 82   // 41*2

// workspace layout (bytes)
#define OFF_EMB      ((size_t)0)            // f16 [5120][64][32]
#define OFF_ENCB     ((size_t)20971520)     // f16 packed 48*512
#define OFF_ENCBIAS  ((size_t)21020672)     // f32 [256]
#define OFF_ENCOUT   ((size_t)21021696)     // f32 [5120][64]
#define OFF_DECXB    ((size_t)22332416)     // f16 packed 128*512
#define OFF_DECHB    ((size_t)22463488)     // f16 packed 128*512
#define OFF_DECBIAS  ((size_t)22594560)     // f32 [512]
#define OFF_DECIN    ((size_t)22596608)     // f16 [4096][128]
#define OFF_XG       ((size_t)23645184)     // f32 [4096][512]

typedef __attribute__((ext_vector_type(8)))  _Float16 h8;
typedef __attribute__((ext_vector_type(16))) _Float16 h16;
typedef __attribute__((ext_vector_type(8)))  float    f8;

typedef unsigned int u32x4 __attribute__((ext_vector_type(4)));
typedef int          i32x8 __attribute__((ext_vector_type(8)));
typedef int          i32x4 __attribute__((ext_vector_type(4)));

__device__ __forceinline__ h16 cat16(h8 lo, h8 hi) {
  h16 r;
#pragma unroll
  for (int e = 0; e < 8; ++e) { r[e] = lo[e]; r[e + 8] = hi[e]; }
  return r;
}

// A fragment (16x32 f16). base points at (row, kbase) in a row-major f16 matrix.
// Lanes 0-15: K {0..15}; lanes 16-31: K {16..31}.
__device__ __forceinline__ h16 afrag(const _Float16* base, bool hiHalf) {
  const h8* p0 = (const h8*)(base + (hiHalf ? 16 : 0));
  const h8* p1 = (const h8*)(base + 8 + (hiHalf ? 16 : 0));
  return cat16(*p0, *p1);
}

// B fragment (32x16 f16) from pre-packed blob: 32 lanes x 16 contiguous halfs.
__device__ __forceinline__ h16 bfrag(const _Float16* base) {
  const h8* p = (const h8*)base;
  return cat16(p[0], p[1]);
}

__device__ __forceinline__ f8 wmma32(h16 a, h16 b, f8 c) {
  return __builtin_amdgcn_wmma_f32_16x16x32_f16(false, a, false, b, (short)0, c, false, false);
}

__device__ __forceinline__ float sigf(float x) { return 1.0f / (1.0f + __expf(-x)); }
__device__ __forceinline__ float lrelu(float x) { return x > 0.0f ? x : 0.1f * x; }

// ---------------------------------------------------------------------------
// Tensor Data Mover: bulk 2-D copy global f16 -> LDS (CDNA5 ISA)
// D# group0: count=1 | lds_addr | global_addr[56:0] | type=2
// D# group1: data_size=2B, tensor_dim0=rowElems, tensor_dim1=rows,
//            tile_dim0=rowElems (<=65535), tile_dim1=rows, dim0_stride=rowElems
// Issued by a single wave (TDM ignores EXEC); completion via TENSORcnt.
// Toolchain here uses the 6-arg builtin form:
//   (u32x4 g0, i32x8 g1, i32x4 g2, i32x4 g3, i32x8, i32 cpol)
// ---------------------------------------------------------------------------
#if __has_builtin(__builtin_amdgcn_tensor_load_to_lds) && \
    __has_builtin(__builtin_amdgcn_s_wait_tensorcnt)
#define USE_TDM 1
#else
#define USE_TDM 0
#endif

#if USE_TDM
__device__ __forceinline__ void tdm_load_2d(unsigned ldsOff, const void* gptr,
                                            unsigned rowElems, unsigned rows) {
  unsigned long long ga = (unsigned long long)(size_t)gptr;
  u32x4 g0;
  g0[0] = 1u;                                               // count=1 (user D#)
  g0[1] = ldsOff;                                           // lds_addr (bytes)
  g0[2] = (unsigned)(ga & 0xFFFFFFFFu);                     // global_addr lo
  g0[3] = (unsigned)((ga >> 32) & 0x01FFFFFFu) | (2u << 30);// addr hi | type=2
  i32x8 g1;
  g1[0] = (int)(1u << 16);                                  // data_size=1 (2B)
  g1[1] = (int)((rowElems & 0xFFFFu) << 16);                // tensor_dim0 lo16
  g1[2] = (int)(((rowElems >> 16) & 0xFFFFu) |              // tensor_dim0 hi16
                ((rows & 0xFFFFu) << 16));                  // tensor_dim1 lo16
  g1[3] = (int)(((rows >> 16) & 0xFFFFu) |                  // tensor_dim1 hi16
                ((rowElems & 0xFFFFu) << 16));              // tile_dim0
  g1[4] = (int)(rows & 0xFFFFu);                            // tile_dim1 (tile_dim2=0)
  g1[5] = (int)rowElems;                                    // dim0_stride lo32
  g1[6] = 0;
  g1[7] = 0;
  i32x4 z4 = {0, 0, 0, 0};
  i32x8 z8 = {0, 0, 0, 0, 0, 0, 0, 0};
  __builtin_amdgcn_tensor_load_to_lds(g0, g1, z4, z4, z8, 0);
}
#endif

// stage f16 elements from global into LDS (TDM path + fallback)
__device__ __forceinline__ void stage_weights(_Float16* ldsDst, const _Float16* gSrc,
                                              unsigned rowElems, unsigned rows) {
#if USE_TDM
  if (threadIdx.x == 0)
    tdm_load_2d((unsigned)(size_t)ldsDst, (const void*)gSrc, rowElems, rows);
  __builtin_amdgcn_s_wait_tensorcnt(0);
#else
  unsigned total = rowElems * rows;
  const uint4* src = (const uint4*)gSrc;
  uint4* dst = (uint4*)ldsDst;
  for (unsigned i = threadIdx.x; i < total / 8u; i += blockDim.x) dst[i] = src[i];
#endif
}

// ---------------------------------------------------------------------------
// Kernel 1: pack weights into WMMA-B layouts + fold biases
// ---------------------------------------------------------------------------
__global__ void pack_kernel(const float* WihE, const float* WhhE,
                            const float* bihE, const float* bhhE,
                            const float* WihD, const float* WhhD,
                            const float* bihD, const float* bhhD,
                            _Float16* encB, _Float16* decXB, _Float16* decHB,
                            float* encBias, float* decBias) {
  int idx = blockIdx.x * blockDim.x + threadIdx.x;
  if (idx < 24576) {                               // encoder B: 16 ntiles x 3 ktiles
    int tile = idx >> 9, r = idx & 511, l = r >> 4, e = r & 15;
    int nb = tile / 3, kb = tile - nb * 3;
    int n = l & 15;
    int k = kb * 32 + ((l >> 4) << 4) + e;         // K = [x(32) | h(64)]
    float w = (k < 32) ? WihE[(nb * 16 + n) * 32 + k]
                       : WhhE[(nb * 16 + n) * 64 + (k - 32)];
    encB[idx] = (_Float16)w;
  } else if (idx < 24576 + 65536) {                // decoder x-weights: 32 x 4
    int id = idx - 24576;
    int tile = id >> 9, r = id & 511, l = r >> 4, e = r & 15;
    int nb = tile >> 2, kb = tile & 3;
    int n = l & 15;
    int k = kb * 32 + ((l >> 4) << 4) + e;
    decXB[id] = (_Float16)WihD[(nb * 16 + n) * 128 + k];
  } else if (idx < 24576 + 131072) {               // decoder h-weights: 32 x 4
    int id = idx - 24576 - 65536;
    int tile = id >> 9, r = id & 511, l = r >> 4, e = r & 15;
    int nb = tile >> 2, kb = tile & 3;
    int n = l & 15;
    int k = kb * 32 + ((l >> 4) << 4) + e;
    decHB[id] = (_Float16)WhhD[(nb * 16 + n) * 128 + k];
  } else if (idx < 24576 + 131072 + 256) {
    int g = idx - (24576 + 131072);
    encBias[g] = bihE[g] + bhhE[g];
  } else if (idx < 24576 + 131072 + 768) {
    int g = idx - (24576 + 131072 + 256);
    decBias[g] = bihD[g] + bhhD[g];
  }
}

// ---------------------------------------------------------------------------
// Kernel 2: input embedding -> f16  emb[seq=b*5+a][t][32]
// ---------------------------------------------------------------------------
__global__ void embed_kernel(const float* pose_tags, const float* pose_nghs,
                             const float* We, const float* be, _Float16* emb) {
  int gid = blockIdx.x * blockDim.x + threadIdx.x;
  if (gid >= 5120 * 64) return;
  int seq = gid >> 6, t = gid & 63;
  int b = seq / 5, a = seq - b * 5;
  float x0, x1;
  if (a == 0) {
    size_t base = ((size_t)b * 64 + t) * 2;
    x0 = pose_tags[base]; x1 = pose_tags[base + 1];
  } else {
    size_t base = (((size_t)b * 4 + (a - 1)) * 64 + t) * 2;
    x0 = pose_nghs[base]; x1 = pose_nghs[base + 1];
  }
#pragma unroll
  for (int k = 0; k < 32; ++k) {
    float z = We[k * 2] * x0 + We[k * 2 + 1] * x1 + be[k];
    emb[(size_t)gid * 32 + k] = (_Float16)lrelu(z);
  }
}

// ---------------------------------------------------------------------------
// Kernel 3: encoder LSTM (WMMA). 128 thr = 4 waves, wave = one 16-seq M-tile.
// Per step: gates(16x256) = [x_t|h] (16x96) @ B(96x256) + bias.
// Weights staged 48 KB global->LDS via TDM once per workgroup.
// ---------------------------------------------------------------------------
__global__ __launch_bounds__(128) void enc_lstm_kernel(const _Float16* emb,
                                                       const _Float16* Bw,
                                                       const float* bias,
                                                       float* encOut) {
  __shared__ __align__(16) _Float16 sB[48 * 512];   // 48 KB staged weights
  __shared__ __align__(16) _Float16 hl[64 * 64];    // 8 KB h relay (f16)

  int tid = threadIdx.x;
  stage_weights(sB, Bw, 24576u, 1u);
  for (int i = tid; i < 64 * 64; i += 128) hl[i] = (_Float16)0.0f;

  int wave = tid >> 5, lane = tid & 31;
  int n = lane & 15;
  bool hiL = lane >= 16;
  int mt = wave;                      // wave's M-tile within block
  int seqbase = blockIdx.x * 64 + mt * 16;

  float biasv[16];
#pragma unroll
  for (int a = 0; a < 16; ++a) biasv[a] = bias[a * 16 + n];

  f8 cst[4];
#pragma unroll
  for (int q = 0; q < 4; ++q)
#pragma unroll
    for (int i = 0; i < 8; ++i) cst[q][i] = 0.0f;

  __syncthreads();

#pragma unroll 1
  for (int t = 0; t < 64; ++t) {
    // A fragments: x_t from global emb (K 0..31), h from LDS (K 32..95)
    const _Float16* ax = emb + (((size_t)(seqbase + n)) * 64 + t) * 32;
    h16 a0 = afrag(ax, hiL);
    const _Float16* ah = hl + (mt * 16 + n) * 64;
    h16 a1 = afrag(ah, hiL);
    h16 a2 = afrag(ah + 32, hiL);

    f8 accs[16];
#pragma unroll
    for (int a = 0; a < 16; ++a) {
      f8 c;
#pragma unroll
      for (int i = 0; i < 8; ++i) c[i] = biasv[a];
      c = wmma32(a0, bfrag(sB + (a * 3 + 0) * 512 + lane * 16), c);
      c = wmma32(a1, bfrag(sB + (a * 3 + 1) * 512 + lane * 16), c);
      c = wmma32(a2, bfrag(sB + (a * 3 + 2) * 512 + lane * 16), c);
      accs[a] = c;
    }

    // gate fusion (lane-local: i/f/g/o share (lane, vgpr) slots across tiles)
    float hv[4][8];
#pragma unroll
    for (int q = 0; q < 4; ++q)
#pragma unroll
      for (int i = 0; i < 8; ++i) {
        float ig = accs[q][i], fg = accs[4 + q][i];
        float gg = accs[8 + q][i], og = accs[12 + q][i];
        float c2 = sigf(fg) * cst[q][i] + sigf(ig) * tanhf(gg);
        cst[q][i] = c2;
        hv[q][i] = sigf(og) * tanhf(c2);
      }

    __syncthreads();   // everyone done reading hl this step
#pragma unroll
    for (int q = 0; q < 4; ++q)
#pragma unroll
      for (int i = 0; i < 8; ++i) {
        int m = mt * 16 + i + (hiL ? 8 : 0);
        hl[m * 64 + q * 16 + n] = (_Float16)hv[q][i];
      }
    __syncthreads();   // h published

    if (t == 63) {
#pragma unroll
      for (int q = 0; q < 4; ++q)
#pragma unroll
        for (int i = 0; i < 8; ++i)
          encOut[((size_t)seqbase + i + (hiL ? 8 : 0)) * 64 + q * 16 + n] = hv[q][i];
    }
  }
}

// ---------------------------------------------------------------------------
// Kernel 4: social pooling + attention + prb head + dec_in build. 1 block / batch.
// ---------------------------------------------------------------------------
__global__ __launch_bounds__(128) void attn_kernel(
    const float* encOut, const float* pose_tags, const float* pose_nghs,
    const float* Wq, const float* bq, const float* Wk, const float* bk,
    const float* Wv, const float* bv, const float* Wf, const float* bf,
    const float* W1, const float* b1, const float* W2, const float* b2,
    _Float16* decIn, float* outPrb) {
  int b = blockIdx.x, tid = threadIdx.x;
  __shared__ float enc[5][64];
  __shared__ int   owner[LCELL];
  __shared__ float qv[128], kn[5][128], vn[5][128];
  __shared__ float maskp[5][4];
  __shared__ float logits[4][LCELL];
  __shared__ float mhp[128], mhpo[32], z1[64], prb[4];

  for (int i = tid; i < 5 * 64; i += 128) enc[i >> 6][i & 63] = encOut[(size_t)b * 320 + i];
  if (tid < LCELL) owner[tid] = -1;
  __syncthreads();

  if (tid == 0) {  // later neighbors overwrite earlier (matches sequential where)
    float px = pose_tags[((size_t)b * 64 + 63) * 2];
    for (int j = 0; j < 4; ++j) {
      size_t base = (((size_t)b * 4 + j) * 64 + 63) * 2;
      float x = pose_nghs[base], y = pose_nghs[base + 1];
      bool valid = fabsf(x + y) >= 1e-4f;
      int ix = 20 + (int)rintf((x - px) * 0.25f);   // round-half-even like jnp.round
      int iy = (y < 0.0f) ? 0 : 1;
      if (valid && ix >= 0 && ix < 41) owner[ix * 2 + iy] = j;
    }
  }
  // q = enc0 @ Wq^T + bq
  {
    float s = bq[tid];
    for (int k = 0; k < 64; ++k) s += enc[0][k] * Wq[tid * 64 + k];
    qv[tid] = s;
  }
  // k/v prototypes: 4 neighbors + empty(=bias only, exact reference semantics)
  for (int idx = tid; idx < 5 * 128; idx += 128) {
    int p = idx >> 7, r = idx & 127;
    float kv = bk[r], vv = bv[r];
    if (p < 4) {
      for (int k = 0; k < 64; ++k) {
        kv += enc[1 + p][k] * Wk[r * 64 + k];
        vv += enc[1 + p][k] * Wv[r * 64 + k];
      }
    }
    kn[p][r] = kv; vn[p][r] = vv;
  }
  __syncthreads();
  if (tid < 20) {
    int p = tid >> 2, h = tid & 3;
    float s = 0.0f;
    for (int d = 0; d < 32; ++d) s += fabsf(kn[p][h * 32 + d]);
    maskp[p][h] = (s < 1e-10f) ? 1.0f : 0.0f;
  }
  __syncthreads();
  for (int idx = tid; idx < 4 * LCELL; idx += 128) {
    int h = idx / LCELL, l = idx - h * LCELL;
    int p = owner[l] >= 0 ? owner[l] : 4;
    float s = 0.0f;
    for (int d = 0; d < 32; ++d) s += qv[h * 32 + d] * kn[p][h * 32 + d];
    logits[h][l] = (maskp[p][h] > 0.0f) ? -1e10f : s * 0.17677669529663687f;
  }
  __syncthreads();
  if (tid < 4) {  // per-head softmax over 82 cells
    float mx = -1e30f;
    for (int l = 0; l < LCELL; ++l) mx = fmaxf(mx, logits[tid][l]);
    float sum = 0.0f;
    for (int l = 0; l < LCELL; ++l) { float e = __expf(logits[tid][l] - mx); logits[tid][l] = e; sum += e; }
    float inv = 1.0f / sum;
    for (int l = 0; l < LCELL; ++l) logits[tid][l] *= inv;
  }
  __syncthreads();
  {
    int h = tid >> 5;
    float s = 0.0f;
    for (int l = 0; l < LCELL; ++l) {
      int p = owner[l] >= 0 ? owner[l] : 4;
      s += logits[h][l] * vn[p][tid];
    }
    mhp[tid] = s;
  }
  __syncthreads();
  if (tid < 32) {
    float s = bf[tid];
    for (int k = 0; k < 128; ++k) s += mhp[k] * Wf[tid * 128 + k];
    mhpo[tid] = s;
  }
  if (tid >= 64) {
    int r = tid - 64;
    float s = b1[r];
    for (int k = 0; k < 128; ++k) s += mhp[k] * W1[r * 128 + k];
    z1[r] = lrelu(s);
  }
  __syncthreads();
  if (tid < 4) {
    float s = b2[tid];
    for (int k = 0; k < 64; ++k) s += z1[k] * W2[tid * 64 + k];
    prb[tid] = s;
  }
  __syncthreads();
  if (tid == 0) {
    float mx = fmaxf(fmaxf(prb[0], prb[1]), fmaxf(prb[2], prb[3]));
    float e0 = __expf(prb[0] - mx), e1 = __expf(prb[1] - mx);
    float e2 = __expf(prb[2] - mx), e3 = __expf(prb[3] - mx);
    float inv = 1.0f / (e0 + e1 + e2 + e3);
    outPrb[b * 4 + 0] = e0 * inv; outPrb[b * 4 + 1] = e1 * inv;
    outPrb[b * 4 + 2] = e2 * inv; outPrb[b * 4 + 3] = e3 * inv;
  }
  // dec_in[b*4+h][128] = [enc0(64) | mhp_out(32) | mhp[h*32..](32)]
  for (int idx = tid; idx < 512; idx += 128) {
    int h = idx >> 7, c = idx & 127;
    float v = (c < 64) ? enc[0][c] : (c < 96 ? mhpo[c - 64] : mhp[h * 32 + (c - 96)]);
    decIn[((size_t)b * 4 + h) * 128 + c] = (_Float16)v;
  }
}

// ---------------------------------------------------------------------------
// Kernel 5: Xg = dec_in @ W_ih_d^T + (b_ih+b_hh)  (4096x512, K=128), WMMA
// ---------------------------------------------------------------------------
__global__ __launch_bounds__(128) void xg_gemm_kernel(const _Float16* din,
                                                      const _Float16* Bw,
                                                      const float* bias,
                                                      float* Xg) {
  int tid = threadIdx.x, wave = tid >> 5, lane = tid & 31;
  int n = lane & 15;
  bool hiL = lane >= 16;
  int mbase = blockIdx.x * 16;

  f8 acc[8];
#pragma unroll
  for (int t2 = 0; t2 < 8; ++t2) {
    float bv = bias[(wave * 8 + t2) * 16 + n];
#pragma unroll
    for (int i = 0; i < 8; ++i) acc[t2][i] = bv;
  }
#pragma unroll
  for (int kb = 0; kb < 4; ++kb) {
    h16 a = afrag(din + (size_t)(mbase + n) * 128 + kb * 32, hiL);
#pragma unroll
    for (int t2 = 0; t2 < 8; ++t2) {
      int nb = wave * 8 + t2;
      acc[t2] = wmma32(a, bfrag(Bw + ((size_t)(nb * 4 + kb) * 512 + lane * 16)), acc[t2]);
    }
  }
#pragma unroll
  for (int t2 = 0; t2 < 8; ++t2)
#pragma unroll
    for (int i = 0; i < 8; ++i)
      Xg[((size_t)mbase + i + (hiL ? 8 : 0)) * 512 + (wave * 8 + t2) * 16 + n] = acc[t2][i];
}

// ---------------------------------------------------------------------------
// Kernel 6: decoder LSTM (WMMA) + fused output projection.
// 128 thr = 4 waves: (M-tile 0/1) x (gate-col half 0/1). Block = 32 seqs.
// Recurrent weights (128 KB f16) staged global->LDS via TDM once per WG;
// every step's 64 B-fragments/wave then come from LDS (no WGP$ thrash).
// ---------------------------------------------------------------------------
__global__ __launch_bounds__(128) void dec_lstm_kernel(const _Float16* Bw,
                                                       const float* Xg,
                                                       const float* Wo,
                                                       const float* bo,
                                                       const float* pose_tags,
                                                       float* out) {
  __shared__ __align__(16) _Float16 sB[128 * 512];  // 128 KB staged weights
  __shared__ __align__(16) _Float16 hl[32 * 128];   // 8 KB h relay
  int tid = threadIdx.x, wave = tid >> 5, lane = tid & 31;
  int n = lane & 15;
  bool hiL = lane >= 16;
  int mt = wave >> 1, jh = wave & 1;
  int mtb = mt * 16;
  int seqb = blockIdx.x * 32;

  stage_weights(sB, Bw, 32768u, 2u);   // 65536 halfs as 2 x 32768 rows (tile_dim0<=65535)

  // hoisted x-gates as persistent C-init registers (reference gate order i|f|g|o)
  f8 xgr[16];
#pragma unroll
  for (int a = 0; a < 16; ++a) {
    int nb = (a >> 2) * 8 + (a & 3) + jh * 4;
#pragma unroll
    for (int i = 0; i < 8; ++i)
      xgr[a][i] = Xg[((size_t)seqb + mtb + i + (hiL ? 8 : 0)) * 512 + nb * 16 + n];
  }
  f8 cst[4];
#pragma unroll
  for (int q = 0; q < 4; ++q)
#pragma unroll
    for (int i = 0; i < 8; ++i) cst[q][i] = 0.0f;

  for (int i = tid; i < 32 * 128; i += 128) hl[i] = (_Float16)0.0f;
  __syncthreads();

#pragma unroll 1
  for (int t = 0; t < PRED; ++t) {
    const _Float16* ah = hl + (mtb + n) * 128;
    h16 a0 = afrag(ah, hiL);
    h16 a1 = afrag(ah + 32, hiL);
    h16 a2 = afrag(ah + 64, hiL);
    h16 a3 = afrag(ah + 96, hiL);

    f8 accs[16];
#pragma unroll
    for (int a = 0; a < 16; ++a) {
      int nb = (a >> 2) * 8 + (a & 3) + jh * 4;
      f8 c = xgr[a];
      c = wmma32(a0, bfrag(sB + ((size_t)(nb * 4 + 0) * 512 + lane * 16)), c);
      c = wmma32(a1, bfrag(sB + ((size_t)(nb * 4 + 1) * 512 + lane * 16)), c);
      c = wmma32(a2, bfrag(sB + ((size_t)(nb * 4 + 2) * 512 + lane * 16)), c);
      c = wmma32(a3, bfrag(sB + ((size_t)(nb * 4 + 3) * 512 + lane * 16)), c);
      accs[a] = c;
    }

    float hv[4][8];
#pragma unroll
    for (int q = 0; q < 4; ++q)
#pragma unroll
      for (int i = 0; i < 8; ++i) {
        float ig = accs[q][i], fg = accs[4 + q][i];
        float gg = accs[8 + q][i], og = accs[12 + q][i];
        float c2 = sigf(fg) * cst[q][i] + sigf(ig) * tanhf(gg);
        cst[q][i] = c2;
        hv[q][i] = sigf(og) * tanhf(c2);
      }

    __syncthreads();   // reads of hl done (incl. last step's out-proj readers)
#pragma unroll
    for (int q = 0; q < 4; ++q)
#pragma unroll
      for (int i = 0; i < 8; ++i) {
        int m = mtb + i + (hiL ? 8 : 0);
        int j = q * 16 + jh * 64 + n;
        hl[m * 128 + j] = (_Float16)hv[q][i];
      }
    __syncthreads();   // h published

    // fused dec_out projection + pose_last add, straight into d_out
    if (tid < 64) {
      int m = tid >> 1, c = tid & 1;
      float s = bo[c];
      for (int k = 0; k < 128; ++k) s += (float)hl[m * 128 + k] * Wo[c * 128 + k];
      int seq = seqb + m;
      int bb = seq >> 2;
      out[(((size_t)seq * PRED) + t) * 2 + c] =
          pose_tags[((size_t)bb * 64 + 63) * 2 + c] + s;
    }
  }
}

// ---------------------------------------------------------------------------
extern "C" void kernel_launch(void* const* d_in, const int* in_sizes, int n_in,
                              void* d_out, int out_size, void* d_ws, size_t ws_size,
                              hipStream_t stream) {
  (void)in_sizes; (void)n_in; (void)out_size; (void)ws_size;
  const float* pose_tags = (const float*)d_in[0];
  const float* pose_nghs = (const float*)d_in[2];
  const float* We    = (const float*)d_in[4];
  const float* be    = (const float*)d_in[5];
  const float* WihE  = (const float*)d_in[6];
  const float* bihE  = (const float*)d_in[7];
  const float* WhhE  = (const float*)d_in[8];
  const float* bhhE  = (const float*)d_in[9];
  const float* Wq    = (const float*)d_in[10];
  const float* bq    = (const float*)d_in[11];
  const float* Wk    = (const float*)d_in[12];
  const float* bk    = (const float*)d_in[13];
  const float* Wv    = (const float*)d_in[14];
  const float* bv    = (const float*)d_in[15];
  const float* Wf    = (const float*)d_in[16];
  const float* bf    = (const float*)d_in[17];
  const float* WihD  = (const float*)d_in[18];
  const float* bihD  = (const float*)d_in[19];
  const float* WhhD  = (const float*)d_in[20];
  const float* bhhD  = (const float*)d_in[21];
  const float* Wo    = (const float*)d_in[22];
  const float* bo    = (const float*)d_in[23];
  const float* W1    = (const float*)d_in[24];
  const float* b1    = (const float*)d_in[25];
  const float* W2    = (const float*)d_in[26];
  const float* b2    = (const float*)d_in[27];

  char* ws = (char*)d_ws;
  _Float16* emb     = (_Float16*)(ws + OFF_EMB);
  _Float16* encB    = (_Float16*)(ws + OFF_ENCB);
  float*    encBias = (float*)   (ws + OFF_ENCBIAS);
  float*    encOut  = (float*)   (ws + OFF_ENCOUT);
  _Float16* decXB   = (_Float16*)(ws + OFF_DECXB);
  _Float16* decHB   = (_Float16*)(ws + OFF_DECHB);
  float*    decBias = (float*)   (ws + OFF_DECBIAS);
  _Float16* decIn   = (_Float16*)(ws + OFF_DECIN);
  float*    Xg      = (float*)   (ws + OFF_XG);

  float* outSeqs = (float*)d_out;
  float* outPrbs = outSeqs + (size_t)Bsz * NHEAD * PRED * 2;

  pack_kernel<<<(156416 + 255) / 256, 256, 0, stream>>>(
      WihE, WhhE, bihE, bhhE, WihD, WhhD, bihD, bhhD,
      encB, decXB, decHB, encBias, decBias);

  embed_kernel<<<(5120 * 64 + 255) / 256, 256, 0, stream>>>(
      pose_tags, pose_nghs, We, be, emb);

  enc_lstm_kernel<<<80, 128, 0, stream>>>(emb, encB, encBias, encOut);

  attn_kernel<<<Bsz, 128, 0, stream>>>(encOut, pose_tags, pose_nghs,
      Wq, bq, Wk, bk, Wv, bv, Wf, bf, W1, b1, W2, b2, decIn, outPrbs);

  xg_gemm_kernel<<<4096 / 16, 128, 0, stream>>>(decIn, decXB, decBias, Xg);

  dec_lstm_kernel<<<4096 / 32, 128, 0, stream>>>(decHB, Xg, Wo, bo,
      pose_tags, outSeqs);
}
